// ModularAttention_23854248362895
// MI455X (gfx1250) — matrix operand order repaired and checked by reading, not compiled, not run on hardware
//
#include <hip/hip_runtime.h>
#include <hip/hip_bf16.h>

namespace {
constexpr int B = 4;
constexpr int S = 4096;
constexpr int D = 1024;
constexpr int KSTEP = 64;  // K-panel depth per staging round
constexpr int LDK = 72;    // LDS row pitch in bf16 (144 B: 16B-aligned, bank-skewed)
}

typedef __bf16 bf16;
typedef __bf16 v16bf __attribute__((ext_vector_type(16)));
typedef __bf16 v8bf  __attribute__((ext_vector_type(8)));
typedef float  v8f   __attribute__((ext_vector_type(8)));

// Pack two f32 -> packed bf16 dword (lowers to v_cvt_pk_bf16_f32).
__device__ inline unsigned pk2(float a, float b) {
  union { unsigned u; bf16 h[2]; } p;
  p.h[0] = (bf16)a;
  p.h[1] = (bf16)b;
  return p.u;
}

// ---- WMMA fragment loads: two contiguous 16-byte LDS reads per fragment ----
// A 16x32 bf16 (row-major tile [m][k]): lane L: m=L&15, hi=L>>4;
//   elems 0..7 = K hi*8..hi*8+7 ; elems 8..15 = K 16+hi*8..16+hi*8+7 (contiguous runs)
__device__ inline v16bf frag_a16(const bf16* row, int hi) {
  union { v16bf v; v8bf h[2]; } f;
  f.h[0] = *(const v8bf*)(row + hi * 8);
  f.h[1] = *(const v8bf*)(row + 16 + hi * 8);
  return f.v;
}
// B 32x16 bf16 stored TRANSPOSED in LDS as [n][k]: lane L: n=L&15, hi=L>>4;
//   elems 0..7 = K hi*8..hi*8+7 ; elems 8..15 = K 16+hi*8..16+hi*8+7 (matches frag_a16)
__device__ inline v16bf frag_bt16(const bf16* row, int hi) {
  union { v16bf v; v8bf h[2]; } f;
  f.h[0] = *(const v8bf*)(row + hi * 8);
  f.h[1] = *(const v8bf*)(row + 16 + hi * 8);
  return f.v;
}

// ---- GEMM 1: scores_t[b,e,d] = (1/32) * sum_s k[b,s,d] * v[b,s,e] (bf16, TRANSPOSED out)
// Block: 256 thr = 8 waves (4 M-waves x 2 N-waves); block tile 128(M=d) x 128(N=e);
// wave tile 32(M) x 64(N) = 2x4 WMMA subtiles; K-dim = S.
__global__ __launch_bounds__(256) void gemm_scores(const float* __restrict__ kmat,
                                                   const float* __restrict__ vmat,
                                                   bf16* __restrict__ scores_t) {
  __shared__ __align__(16) bf16 lds_a[128 * LDK];   // [m=d][k=s]
  __shared__ __align__(16) bf16 lds_bt[128 * LDK];  // [n=e][k=s]
  const int t  = threadIdx.x;
  const int b  = blockIdx.z;
  const int d0 = blockIdx.x * 128;
  const int e0 = blockIdx.y * 128;
  const int wave = t >> 5;
  const int wm = wave >> 1;  // 0..3
  const int wn = wave & 1;   // 0..1
  const int lane = t & 31;
  const int lm = lane & 15;
  const int hi = lane >> 4;
  v8f acc[2][4] = {};
  for (int kk = 0; kk < S; kk += KSTEP) {
    // A panel: k[b, kk..kk+64, d0..d0+128] -> transpose lds_a[d][s].
    // Each task: two float4 (adjacent s-rows, same 4 d-cols) -> 4 packed b32 stores.
#pragma unroll
    for (int j = 0; j < 4; ++j) {
      const int idx = j * 256 + t;     // 1024 tasks (8 f32 each)
      const int sp  = idx >> 5;        // s-pair 0..31 -> rows 2sp, 2sp+1
      const int mc  = (idx & 31) * 4;  // 0..124
      const float4 x = *(const float4*)(kmat + ((size_t)(b * S + kk + 2 * sp)) * D + d0 + mc);
      const float4 y = *(const float4*)(kmat + ((size_t)(b * S + kk + 2 * sp + 1)) * D + d0 + mc);
      *(unsigned*)&lds_a[(mc + 0) * LDK + 2 * sp] = pk2(x.x, y.x);
      *(unsigned*)&lds_a[(mc + 1) * LDK + 2 * sp] = pk2(x.y, y.y);
      *(unsigned*)&lds_a[(mc + 2) * LDK + 2 * sp] = pk2(x.z, y.z);
      *(unsigned*)&lds_a[(mc + 3) * LDK + 2 * sp] = pk2(x.w, y.w);
    }
    // B panel: v[b, kk..kk+64, e0..e0+128] -> transpose lds_bt[e][s], same scheme.
#pragma unroll
    for (int j = 0; j < 4; ++j) {
      const int idx = j * 256 + t;
      const int sp  = idx >> 5;
      const int nc  = (idx & 31) * 4;
      const float4 x = *(const float4*)(vmat + ((size_t)(b * S + kk + 2 * sp)) * D + e0 + nc);
      const float4 y = *(const float4*)(vmat + ((size_t)(b * S + kk + 2 * sp + 1)) * D + e0 + nc);
      *(unsigned*)&lds_bt[(nc + 0) * LDK + 2 * sp] = pk2(x.x, y.x);
      *(unsigned*)&lds_bt[(nc + 1) * LDK + 2 * sp] = pk2(x.y, y.y);
      *(unsigned*)&lds_bt[(nc + 2) * LDK + 2 * sp] = pk2(x.z, y.z);
      *(unsigned*)&lds_bt[(nc + 3) * LDK + 2 * sp] = pk2(x.w, y.w);
    }
    __syncthreads();
#pragma unroll
    for (int ks = 0; ks < KSTEP; ks += 32) {
      v16bf af[2], bfr[4];
#pragma unroll
      for (int mi = 0; mi < 2; ++mi)
        af[mi] = frag_a16(&lds_a[(wm * 32 + mi * 16 + lm) * LDK + ks], hi);
#pragma unroll
      for (int ni = 0; ni < 4; ++ni)
        bfr[ni] = frag_bt16(&lds_bt[(wn * 64 + ni * 16 + lm) * LDK + ks], hi);
#pragma unroll
      for (int mi = 0; mi < 2; ++mi)
#pragma unroll
        for (int ni = 0; ni < 4; ++ni)
          acc[mi][ni] = __builtin_amdgcn_wmma_f32_16x16x32_bf16(
              false, af[mi], false, bfr[ni], (short)0, acc[mi][ni], false, false);
    }
    __syncthreads();
  }
  // Store transposed: scores_t[b][e][d]; each lane's 8 acc elems contiguous in d -> b128.
#pragma unroll
  for (int mi = 0; mi < 2; ++mi)
#pragma unroll
    for (int ni = 0; ni < 4; ++ni) {
      const int ng = e0 + wn * 64 + ni * 16 + lm;
      const int mg = d0 + wm * 32 + mi * 16 + 8 * hi;  // elem i -> mg + i
      union { uint4 u4; bf16 h[8]; } p;
#pragma unroll
      for (int i = 0; i < 8; ++i) p.h[i] = (bf16)(acc[mi][ni][i] * 0.03125f);
      *(uint4*)(scores_t + ((size_t)b * D + ng) * D + mg) = p.u4;
    }
}

// ---- q normalization chain (deterministic 2-stage reductions over S) ----
__global__ __launch_bounds__(256) void exp_partial(const float* __restrict__ q,
                                                   float* __restrict__ partial) {
  const int d = blockIdx.x * 256 + threadIdx.x;
  const int c = blockIdx.y;
  const int b = blockIdx.z;
  const float* p = q + ((size_t)b * S + (size_t)c * 512) * D + d;
  float acc = 0.f;
  for (int s = 0; s < 512; ++s) acc += __expf(p[(size_t)s * D]);
  partial[((size_t)b * 8 + c) * D + d] = acc;
}

__global__ __launch_bounds__(256) void combine_denom(const float* __restrict__ partial,
                                                     float* __restrict__ denom) {
  const int i = blockIdx.x * 256 + threadIdx.x;  // B*D
  const int b = i >> 10, d = i & (D - 1);
  float acc = 1.0f;  // denom = 1 + sum exp(q)
#pragma unroll
  for (int c = 0; c < 8; ++c) acc += partial[((size_t)b * 8 + c) * D + d];
  denom[i] = acc;
}

__global__ __launch_bounds__(256) void softexp_partial(const float* __restrict__ q,
                                                       const float* __restrict__ denom,
                                                       float* __restrict__ partial) {
  const int d = blockIdx.x * 256 + threadIdx.x;
  const int c = blockIdx.y;
  const int b = blockIdx.z;
  const float dd = denom[b * D + d];
  const float* p = q + ((size_t)b * S + (size_t)c * 512) * D + d;
  float acc = 0.f;
  // qn = exp(q)/denom in (0,1): softmax max-subtraction safely skipped
  for (int s = 0; s < 512; ++s) acc += __expf(__expf(p[(size_t)s * D]) / dd);
  partial[((size_t)b * 8 + c) * D + d] = acc;
}

__global__ __launch_bounds__(256) void combine_sum2(const float* __restrict__ partial,
                                                    float* __restrict__ sum2) {
  const int i = blockIdx.x * 256 + threadIdx.x;
  const int b = i >> 10, d = i & (D - 1);
  float acc = 0.0f;
#pragma unroll
  for (int c = 0; c < 8; ++c) acc += partial[((size_t)b * 8 + c) * D + d];
  sum2[i] = acc;
}

__global__ __launch_bounds__(256) void normalize_q(const float* __restrict__ q,
                                                   const float* __restrict__ denom,
                                                   const float* __restrict__ sum2,
                                                   bf16* __restrict__ qbf) {
  const int i = blockIdx.x * 256 + threadIdx.x;  // B*S*D = 2^24
  const int d = i & (D - 1);
  const int b = i >> 22;                         // S*D = 2^22
  const int bd = b * D + d;
  const float qn = __expf(q[i]) / denom[bd];
  qbf[i] = (bf16)(__expf(qn) / sum2[bd]);
}

// ---- GEMM 2: out[b,s,e] = sum_d qn[b,s,d] * scores_t[b,e,d]  (f32 out) ----
// Block tile 128(M=s) x 128(N=e); wave tile 32x64 (2x4 subtiles); K-dim = D.
__global__ __launch_bounds__(256) void gemm_out(const bf16* __restrict__ qn,
                                                const bf16* __restrict__ scores_t,
                                                float* __restrict__ out) {
  __shared__ __align__(16) bf16 lds_a[128 * LDK];   // [m=s][k=d]
  __shared__ __align__(16) bf16 lds_bt[128 * LDK];  // [n=e][k=d]
  const int t  = threadIdx.x;
  const int b  = blockIdx.z;
  const int s0 = blockIdx.x * 128;
  const int e0 = blockIdx.y * 128;
  const int wave = t >> 5;
  const int wm = wave >> 1;
  const int wn = wave & 1;
  const int lane = t & 31;
  const int lm = lane & 15;
  const int hi = lane >> 4;
  v8f acc[2][4] = {};
  for (int kk = 0; kk < D; kk += KSTEP) {
    // A panel: qn[b, s0..s0+128, kk..kk+64] bf16 row-major; uint4 (8xbf16) copies.
#pragma unroll
    for (int j = 0; j < 4; ++j) {
      const int idx = j * 256 + t;    // 1024 uint4
      const int r   = idx >> 3;       // 0..127
      const int c8  = (idx & 7) * 8;  // 0..56
      const uint4 w = *(const uint4*)(qn + ((size_t)b * S + s0 + r) * D + kk + c8);
      *(uint4*)&lds_a[r * LDK + c8] = w;
    }
    // B panel: scores_t[b, e0..e0+128, kk..kk+64] bf16 row-major -> lds_bt[n][k].
#pragma unroll
    for (int j = 0; j < 4; ++j) {
      const int idx = j * 256 + t;    // 1024 uint4
      const int r   = idx >> 3;       // 0..127
      const int c8  = (idx & 7) * 8;
      const uint4 w = *(const uint4*)(scores_t + ((size_t)b * D + e0 + r) * D + kk + c8);
      *(uint4*)&lds_bt[r * LDK + c8] = w;
    }
    __syncthreads();
#pragma unroll
    for (int ks = 0; ks < KSTEP; ks += 32) {
      v16bf af[2], bfr[4];
#pragma unroll
      for (int mi = 0; mi < 2; ++mi)
        af[mi] = frag_a16(&lds_a[(wm * 32 + mi * 16 + lm) * LDK + ks], hi);
#pragma unroll
      for (int ni = 0; ni < 4; ++ni)
        bfr[ni] = frag_bt16(&lds_bt[(wn * 64 + ni * 16 + lm) * LDK + ks], hi);
#pragma unroll
      for (int mi = 0; mi < 2; ++mi)
#pragma unroll
        for (int ni = 0; ni < 4; ++ni)
          acc[mi][ni] = __builtin_amdgcn_wmma_f32_16x16x32_bf16(
              false, af[mi], false, bfr[ni], (short)0, acc[mi][ni], false, false);
    }
    __syncthreads();
  }
#pragma unroll
  for (int mi = 0; mi < 2; ++mi)
#pragma unroll
    for (int ni = 0; ni < 4; ++ni) {
      const int ng = e0 + wn * 64 + ni * 16 + lm;
      const int mg = s0 + wm * 32 + mi * 16 + 8 * hi;
#pragma unroll
      for (int i = 0; i < 8; ++i)
        out[((size_t)b * S + mg + i) * D + ng] = acc[mi][ni][i];
    }
}

extern "C" void kernel_launch(void* const* d_in, const int* in_sizes, int n_in,
                              void* d_out, int out_size, void* d_ws, size_t ws_size,
                              hipStream_t stream) {
  (void)in_sizes; (void)n_in; (void)out_size; (void)ws_size;
  const float* kmat = (const float*)d_in[0];  // k: [B,S,D] f32
  const float* qmat = (const float*)d_in[1];  // q: [B,S,D] f32
  const float* vmat = (const float*)d_in[2];  // v: [B,S,D] f32
  float* out = (float*)d_out;                 // [B,S,D] f32

  // Workspace layout:
  //   scores_t : B*D*D bf16 (8 MB, stored [b][e][d])
  //   qbf      : B*S*D bf16 (33.5 MB)
  //   denom    : B*D f32 ; sum2 : B*D f32 ; partial : B*8*D f32
  bf16* scores_t = (bf16*)d_ws;
  bf16* qbf      = scores_t + (size_t)B * D * D;
  float* denom   = (float*)(qbf + (size_t)B * S * D);
  float* sum2    = denom + (size_t)B * D;
  float* partial = sum2 + (size_t)B * D;

  // scores^T = (K^T V)^T / sqrt(D)
  gemm_scores<<<dim3(D / 128, D / 128, B), 256, 0, stream>>>(kmat, vmat, scores_t);

  // qn = softmax_s(exp(q) / (1 + sum_s exp(q))) -> bf16
  exp_partial<<<dim3(D / 256, 8, B), 256, 0, stream>>>(qmat, partial);
  combine_denom<<<dim3((B * D) / 256), 256, 0, stream>>>(partial, denom);
  softexp_partial<<<dim3(D / 256, 8, B), 256, 0, stream>>>(qmat, denom, partial);
  combine_sum2<<<dim3((B * D) / 256), 256, 0, stream>>>(partial, sum2);
  normalize_q<<<dim3((B * S * D) / 256), 256, 0, stream>>>(qmat, denom, sum2, qbf);

  // out = qn @ scores
  gemm_out<<<dim3(S / 128, D / 128, B), 256, 0, stream>>>(qbf, scores_t, out);
}
